// Stacked1DSelfAttention_81372450390506
// MI455X (gfx1250) — compile-verified, matrix operand order checked by reading, not run on hardware
//
#include <hip/hip_runtime.h>
#include <math.h>

// ---------------- problem constants ----------------
constexpr int Bc   = 4;
constexpr int Cc   = 192;
constexpr int Nc   = 160;
constexpr int Mc   = 160;
constexpr int Hc   = 4;
constexpr int DHc  = 32;
constexpr int HIDc = 128;   // Hc*DHc
constexpr int NMEMc = 4;
#define SCALE_F 0.17677669529663687f   // 1/sqrt(32)

typedef __attribute__((ext_vector_type(16))) _Float16 v16h;
typedef __attribute__((ext_vector_type(8)))  float    v8f;

// =====================================================================
// Kernel 1: per-pixel inverse channel norm of x:  invp[b,n,m]
// =====================================================================
__global__ __launch_bounds__(256) void pixel_invnorm(const float* __restrict__ x,
                                                     float* __restrict__ invp)
{
    const int idx = blockIdx.x * blockDim.x + threadIdx.x;
    const int NM = Nc * Mc;
    if (idx >= Bc * NM) return;
    const int b = idx / NM;
    const int p = idx - b * NM;
    const float* xp = x + (size_t)b * Cc * NM + p;
    float s = 0.f;
    for (int c = 0; c < Cc; ++c) {
        float v = xp[(size_t)c * NM];
        s += v * v;
    }
    invp[idx] = 1.0f / fmaxf(sqrtf(s), 1e-12f);
}

// =====================================================================
// Kernel 2: per (b,c) plane reductions.
//  row_ft[b,c,n] = mean_m x          col_ft[b,c,m] = mean_n x
//  Scol[b,c,n]   = sum_m x*invp      Srow[b,c,m]   = sum_n x*invp
// block = 160 threads, grid = (C, B)
// =====================================================================
__global__ __launch_bounds__(160) void plane_reduce(const float* __restrict__ x,
                                                    const float* __restrict__ invp,
                                                    float* __restrict__ row_ft,
                                                    float* __restrict__ col_ft,
                                                    float* __restrict__ Srow,
                                                    float* __restrict__ Scol)
{
    const int b = blockIdx.y, c = blockIdx.x;
    const float* xp = x + ((size_t)b * Cc + c) * Nc * Mc;
    const float* ip = invp + (size_t)b * Nc * Mc;
    const int t = threadIdx.x;
    if (t < Nc) {   // row t: reduce over m
        float s = 0.f, sw = 0.f;
        for (int m = 0; m < Mc; ++m) {
            float v = xp[t * Mc + m];
            s += v;
            sw += v * ip[t * Mc + m];
        }
        row_ft[((size_t)b * Cc + c) * Nc + t] = s * (1.0f / Mc);
        Scol  [((size_t)b * Cc + c) * Nc + t] = sw;
    }
    if (t < Mc) {   // col t: reduce over n
        float s = 0.f, sw = 0.f;
        for (int n = 0; n < Nc; ++n) {
            float v = xp[n * Mc + t];
            s += v;
            sw += v * ip[n * Mc + t];
        }
        col_ft[((size_t)b * Cc + c) * Mc + t] = s * (1.0f / Nc);
        Srow  [((size_t)b * Cc + c) * Mc + t] = sw;
    }
}

// =====================================================================
// Kernel 3: inverse channel norm of a (B,C,P) feature per column p
// =====================================================================
__global__ __launch_bounds__(128) void feat_invnorm(const float* __restrict__ f,
                                                    float* __restrict__ inv,
                                                    int P, int total)
{
    const int idx = blockIdx.x * blockDim.x + threadIdx.x;
    if (idx >= total) return;
    const int b = idx / P;
    const int p = idx - b * P;
    const float* fp = f + (size_t)b * Cc * P + p;
    float s = 0.f;
    for (int c = 0; c < Cc; ++c) {
        float v = fp[(size_t)c * P];
        s += v * v;
    }
    inv[idx] = 1.0f / fmaxf(sqrtf(s), 1e-12f);
}

// =====================================================================
// Kernel 4: generic WMMA GEMM (one wave = one 16x16 output tile)
//   Out[b,m,p] = sum_k (W[m,k]*cs[k]) * (X[b,k,p] * ps[b,p] * xscale) + bias[m]
// Compile-time specialization removes all null-pointer branches.
// A/cs loads are float4 (global_load_b128); B loads coalesce across lanes.
// Fragment layouts per CDNA5 ISA 7.12.2 (16-bit A 16x32; B row-striped).
// =====================================================================
template <bool HAS_CS, bool HAS_PS, bool HAS_BIAS>
__global__ __launch_bounds__(32) void gemm_wmma(const float* __restrict__ W, int ldW,
                                                const float* __restrict__ X,
                                                const float* __restrict__ cs,
                                                const float* __restrict__ ps,
                                                const float* __restrict__ bias,
                                                float xscale,
                                                float* __restrict__ Out,
                                                int Mdim, int Kdim, int Nn)
{
    const int b  = blockIdx.z;
    const int mt = blockIdx.y;
    const int nt = blockIdx.x;
    const int lane = threadIdx.x;
    const int nl = lane & 15;
    const int k0 = (lane & 16) ? 8 : 0;

    const int mrow = (mt << 4) + nl;           // A-matrix row held by this lane
    const int ncol = (nt << 4) + nl;           // B/D column held by this lane
    const float* Xb = X + (size_t)b * Kdim * Nn;
    const float  psn = (HAS_PS ? ps[(size_t)b * Nn + ncol] : 1.0f) * xscale;
    const float* Wrow = W + (size_t)mrow * ldW;

    v8f acc = {};
    for (int kk = 0; kk < Kdim; kk += 32) {
        const int ka = kk + k0;       // multiple of 4 -> 16B aligned
        const int kb = ka + 16;
        const float4 w0 = *(const float4*)(Wrow + ka);
        const float4 w1 = *(const float4*)(Wrow + ka + 4);
        const float4 w2 = *(const float4*)(Wrow + kb);
        const float4 w3 = *(const float4*)(Wrow + kb + 4);
        float wa[16] = { w0.x, w0.y, w0.z, w0.w, w1.x, w1.y, w1.z, w1.w,
                         w2.x, w2.y, w2.z, w2.w, w3.x, w3.y, w3.z, w3.w };
        if (HAS_CS) {
            const float4 c0 = *(const float4*)(cs + ka);
            const float4 c1 = *(const float4*)(cs + ka + 4);
            const float4 c2 = *(const float4*)(cs + kb);
            const float4 c3 = *(const float4*)(cs + kb + 4);
            const float ca[16] = { c0.x, c0.y, c0.z, c0.w, c1.x, c1.y, c1.z, c1.w,
                                   c2.x, c2.y, c2.z, c2.w, c3.x, c3.y, c3.z, c3.w };
#pragma unroll
            for (int j = 0; j < 16; ++j) wa[j] *= ca[j];
        }
        v16h a, bf;
#pragma unroll
        for (int j = 0; j < 8; ++j) {
            a[j]     = (_Float16)wa[j];
            a[8 + j] = (_Float16)wa[8 + j];
            bf[j]     = (_Float16)(Xb[(size_t)(ka + j) * Nn + ncol] * psn);
            bf[8 + j] = (_Float16)(Xb[(size_t)(kb + j) * Nn + ncol] * psn);
        }
        acc = __builtin_amdgcn_wmma_f32_16x16x32_f16(false, a, false, bf,
                                                     (short)0, acc, false, false);
    }
    const int rbase = (lane >> 4) * 8;
#pragma unroll
    for (int r = 0; r < 8; ++r) {
        const int m = (mt << 4) + rbase + r;
        float v = acc[r];
        if (HAS_BIAS) v += bias[m];
        Out[((size_t)b * Mdim + m) * Nn + ncol] = v;
    }
}

// =====================================================================
// Kernel 5: attention for one (b, h, 16-query tile). One wave per block.
//   Q,K,V are (B, 128, P) row-major (row = h*32+d, col = position).
//   s = (Q^T·K)·SCALE; exact max over [keys, mem]; Z = Σe_m + rep·Σe_k
//   O[b, h*32+d, q] = (e_k·v_sum + e_m·mem_v) / Z
// =====================================================================
#define PADS 168
__global__ __launch_bounds__(32) void attn_wave(const float* __restrict__ Q,
                                                const float* __restrict__ Kf,
                                                const float* __restrict__ V,
                                                const float* __restrict__ memkv,
                                                float* __restrict__ O,
                                                int P, float rep)
{
    __shared__ float s[16][PADS];
    __shared__ float sm[16][NMEMc];
    __shared__ float rowZ[16];

    const int b = blockIdx.z, h = blockIdx.y, qt = blockIdx.x;
    const int lane = threadIdx.x;
    const int nl = lane & 15;
    const int k0 = (lane & 16) ? 8 : 0;
    const size_t base = ((size_t)b * HIDc + h * DHc) * P;
    const float* Qb = Q + base;
    const float* Kb = Kf + base;
    const float* Vb = V + base;
    const int qcol = qt * 16 + nl;

    // A fragment: A[i][d] = Q[b, h*32+d, qt*16+i]   (reused across all key tiles)
    v16h aq;
#pragma unroll
    for (int j = 0; j < 8; ++j) {
        aq[j]     = (_Float16)Qb[(k0 + j) * P + qcol];
        aq[8 + j] = (_Float16)Qb[(16 + k0 + j) * P + qcol];
    }

    // scores vs all key tiles (K = 32 = exactly one WMMA step)
    for (int jt = 0; jt < P / 16; ++jt) {
        const int kcol = jt * 16 + nl;
        v16h bk;
#pragma unroll
        for (int j = 0; j < 8; ++j) {
            bk[j]     = (_Float16)Kb[(k0 + j) * P + kcol];
            bk[8 + j] = (_Float16)Kb[(16 + k0 + j) * P + kcol];
        }
        v8f c = {};
        c = __builtin_amdgcn_wmma_f32_16x16x32_f16(false, aq, false, bk,
                                                   (short)0, c, false, false);
        const int rbase = (lane >> 4) * 8;
#pragma unroll
        for (int r = 0; r < 8; ++r)
            s[rbase + r][jt * 16 + nl] = c[r] * SCALE_F;
    }

    // memory-slot scores (scalar; 16 rows handled by lanes 0..15)
    if (lane < 16) {
#pragma unroll
        for (int jm = 0; jm < NMEMc; ++jm) {
            float acc = 0.f;
            for (int d = 0; d < DHc; ++d)
                acc += Qb[d * P + qt * 16 + lane] * memkv[(h * NMEMc + jm) * DHc + d];
            sm[lane][jm] = acc * SCALE_F;
        }
    }
    __syncthreads();

    // exact softmax stats per row
    if (lane < 16) {
        float mx = -3.4e38f;
        for (int j = 0; j < P; ++j) mx = fmaxf(mx, s[lane][j]);
#pragma unroll
        for (int jm = 0; jm < NMEMc; ++jm) mx = fmaxf(mx, sm[lane][jm]);
        float se = 0.f;
        for (int j = 0; j < P; ++j) {
            float e = expf(s[lane][j] - mx);
            s[lane][j] = e;
            se += e;
        }
        float sem = 0.f;
#pragma unroll
        for (int jm = 0; jm < NMEMc; ++jm) {
            float e = expf(sm[lane][jm] - mx);
            sm[lane][jm] = e;
            sem += e;
        }
        rowZ[lane] = sem + rep * se;
    }
    __syncthreads();

    // out = e_k @ v_sum  (K = P in steps of 32), two 16-wide d-tiles
    const int memv_off = Hc * NMEMc * DHc;   // mem_kv[1]
    for (int nt2 = 0; nt2 < 2; ++nt2) {
        v8f c = {};
        for (int kk = 0; kk < P; kk += 32) {
            // e-matrix row (contiguous in LDS, 16B aligned) and V row (contiguous
            // in global, 16B aligned) -> vector loads
            const float4 e0 = *(const float4*)(&s[nl][kk + k0]);
            const float4 e1 = *(const float4*)(&s[nl][kk + k0 + 4]);
            const float4 e2 = *(const float4*)(&s[nl][kk + 16 + k0]);
            const float4 e3 = *(const float4*)(&s[nl][kk + 16 + k0 + 4]);
            const float* Vrow = Vb + (size_t)(nt2 * 16 + nl) * P;
            const float4 v0 = *(const float4*)(Vrow + kk + k0);
            const float4 v1 = *(const float4*)(Vrow + kk + k0 + 4);
            const float4 v2 = *(const float4*)(Vrow + kk + 16 + k0);
            const float4 v3 = *(const float4*)(Vrow + kk + 16 + k0 + 4);
            const float ea[16] = { e0.x, e0.y, e0.z, e0.w, e1.x, e1.y, e1.z, e1.w,
                                   e2.x, e2.y, e2.z, e2.w, e3.x, e3.y, e3.z, e3.w };
            const float va[16] = { v0.x, v0.y, v0.z, v0.w, v1.x, v1.y, v1.z, v1.w,
                                   v2.x, v2.y, v2.z, v2.w, v3.x, v3.y, v3.z, v3.w };
            v16h ae, bv;
#pragma unroll
            for (int j = 0; j < 16; ++j) {
                ae[j] = (_Float16)ea[j];
                bv[j] = (_Float16)va[j];
            }
            c = __builtin_amdgcn_wmma_f32_16x16x32_f16(false, ae, false, bv,
                                                       (short)0, c, false, false);
        }
        const int rbase = (lane >> 4) * 8;
#pragma unroll
        for (int r = 0; r < 8; ++r) {
            const int i = rbase + r;
            const int d = nt2 * 16 + nl;
            float memc = 0.f;
#pragma unroll
            for (int jm = 0; jm < NMEMc; ++jm)
                memc += sm[i][jm] * memkv[memv_off + (h * NMEMc + jm) * DHc + d];
            O[((size_t)b * HIDc + h * DHc + d) * P + qt * 16 + i] =
                (c[r] + memc) / rowZ[i];
        }
    }
}

// =====================================================================
// Kernel 6: out[b,d,n,m] = Rrow[b,d,n] + Rcol[b,d,m]   (float4 writes)
// =====================================================================
__global__ __launch_bounds__(256) void bcast_add(const float* __restrict__ Rrow,
                                                 const float* __restrict__ Rcol,
                                                 float* __restrict__ out)
{
    const size_t idx4 = (size_t)blockIdx.x * blockDim.x + threadIdx.x;
    const size_t e = idx4 * 4;               // grid sized exactly: no guard needed
    const int m = (int)(e % Mc);
    size_t t = e / Mc;
    const int n = (int)(t % Nc); t /= Nc;
    const int d = (int)(t % Cc);
    const int b = (int)(t / Cc);
    const float  rv = Rrow[((size_t)b * Cc + d) * Nc + n];
    const float4 cv = *(const float4*)(Rcol + ((size_t)b * Cc + d) * Mc + m);
    float4 o;
    o.x = rv + cv.x; o.y = rv + cv.y; o.z = rv + cv.z; o.w = rv + cv.w;
    *(float4*)(out + e) = o;
}

// =====================================================================
// Host launcher
// =====================================================================
extern "C" void kernel_launch(void* const* d_in, const int* in_sizes, int n_in,
                              void* d_out, int out_size, void* d_ws, size_t ws_size,
                              hipStream_t stream)
{
    const float* x     = (const float*)d_in[0];
    const float* gq    = (const float*)d_in[1];
    const float* gk    = (const float*)d_in[2];
    const float* gv    = (const float*)d_in[3];
    const float* Wq    = (const float*)d_in[4];
    const float* Wk    = (const float*)d_in[5];
    const float* Wv    = (const float*)d_in[6];
    const float* memkv = (const float*)d_in[7];
    const float* Wo    = (const float*)d_in[8];
    const float* bo    = (const float*)d_in[9];
    const float* Wr    = (const float*)d_in[10];
    float* out = (float*)d_out;

    // ---- workspace layout (floats); total ~7.0 MB ----
    float* w = (float*)d_ws;
    size_t o = 0;
    float* invp     = w + o; o += (size_t)Bc * Nc * Mc;   // 102400
    float* row_ft   = w + o; o += (size_t)Bc * Cc * Nc;   // 122880
    float* col_ft   = w + o; o += (size_t)Bc * Cc * Mc;
    float* Srow     = w + o; o += (size_t)Bc * Cc * Mc;
    float* Scol     = w + o; o += (size_t)Bc * Cc * Nc;
    float* inv_nrow = w + o; o += (size_t)Bc * Nc;
    float* inv_ncol = w + o; o += (size_t)Bc * Mc;
    float* Qr = w + o; o += (size_t)Bc * HIDc * Nc;       // 81920 each
    float* Kr = w + o; o += (size_t)Bc * HIDc * Mc;
    float* Vr = w + o; o += (size_t)Bc * HIDc * Mc;
    float* Qc = w + o; o += (size_t)Bc * HIDc * Mc;
    float* Kc = w + o; o += (size_t)Bc * HIDc * Nc;
    float* Vc = w + o; o += (size_t)Bc * HIDc * Nc;
    float* Orow = w + o; o += (size_t)Bc * HIDc * Nc;
    float* Ocol = w + o; o += (size_t)Bc * HIDc * Mc;
    float* row_attn = w + o; o += (size_t)Bc * Cc * Nc;
    float* col_attn = w + o; o += (size_t)Bc * Cc * Mc;
    float* Rrow = w + o; o += (size_t)Bc * Cc * Nc;
    float* Rcol = w + o; o += (size_t)Bc * Cc * Mc;
    (void)ws_size; (void)in_sizes; (void)n_in; (void)out_size;

    const float sqrtC = sqrtf((float)Cc);

    // 1) per-pixel channel norms (pass 1 over x)
    pixel_invnorm<<<(Bc * Nc * Mc + 255) / 256, 256, 0, stream>>>(x, invp);

    // 2) plane reductions (pass 2 over x)
    plane_reduce<<<dim3(Cc, Bc), 160, 0, stream>>>(x, invp, row_ft, col_ft, Srow, Scol);

    // 3) feature channel norms
    feat_invnorm<<<(Bc * Nc + 127) / 128, 128, 0, stream>>>(row_ft, inv_nrow, Nc, Bc * Nc);
    feat_invnorm<<<(Bc * Mc + 127) / 128, 128, 0, stream>>>(col_ft, inv_ncol, Mc, Bc * Mc);

    // 4) QKV projections (128x192x160 WMMA GEMMs)
    const dim3 gP(Nc / 16, HIDc / 16, Bc);
    // row branch: q<-row_ft, k<-col_ft, v<-Srow
    gemm_wmma<true, true, false><<<gP, 32, 0, stream>>>(Wq, Cc, row_ft, gq, inv_nrow, nullptr, sqrtC, Qr, HIDc, Cc, Nc);
    gemm_wmma<true, true, false><<<gP, 32, 0, stream>>>(Wk, Cc, col_ft, gk, inv_ncol, nullptr, sqrtC, Kr, HIDc, Cc, Mc);
    gemm_wmma<true, false, false><<<gP, 32, 0, stream>>>(Wv, Cc, Srow, gv, nullptr, nullptr, sqrtC, Vr, HIDc, Cc, Mc);
    // col branch: q<-col_ft, k<-row_ft, v<-Scol
    gemm_wmma<true, true, false><<<gP, 32, 0, stream>>>(Wq, Cc, col_ft, gq, inv_ncol, nullptr, sqrtC, Qc, HIDc, Cc, Mc);
    gemm_wmma<true, true, false><<<gP, 32, 0, stream>>>(Wk, Cc, row_ft, gk, inv_nrow, nullptr, sqrtC, Kc, HIDc, Cc, Nc);
    gemm_wmma<true, false, false><<<gP, 32, 0, stream>>>(Wv, Cc, Scol, gv, nullptr, nullptr, sqrtC, Vc, HIDc, Cc, Nc);

    // 5) attention (one wave per (b,h,q-tile))
    const dim3 gA(Nc / 16, Hc, Bc);
    attn_wave<<<gA, 32, 0, stream>>>(Qr, Kr, Vr, memkv, Orow, Nc, (float)Nc);
    attn_wave<<<gA, 32, 0, stream>>>(Qc, Kc, Vc, memkv, Ocol, Mc, (float)Mc);

    // 6) output projections (192x128x160) with bias bo
    const dim3 gO(Nc / 16, Cc / 16, Bc);
    gemm_wmma<false, false, true><<<gO, 32, 0, stream>>>(Wo, HIDc, Orow, nullptr, nullptr, bo, 1.0f, row_attn, Cc, HIDc, Nc);
    gemm_wmma<false, false, true><<<gO, 32, 0, stream>>>(Wo, HIDc, Ocol, nullptr, nullptr, bo, 1.0f, col_attn, Cc, HIDc, Mc);

    // 7) Wr projections (192x192x160); Wr is (C, 2C): Wr1 = Wr[:, :C], Wr2 = Wr[:, C:]
    gemm_wmma<false, false, false><<<gO, 32, 0, stream>>>(Wr, 2 * Cc, row_attn, nullptr, nullptr, nullptr, 1.0f, Rrow, Cc, Cc, Nc);
    gemm_wmma<false, false, false><<<gO, 32, 0, stream>>>(Wr + Cc, 2 * Cc, col_attn, nullptr, nullptr, nullptr, 1.0f, Rcol, Cc, Cc, Mc);

    // 8) broadcast add -> final output (78.6 MB, float4 stores)
    const int total4 = (Bc * Cc * Nc * Mc) / 4;       // 4,915,200
    bcast_add<<<total4 / 256, 256, 0, stream>>>(Rrow, Rcol, out);
}
